// MyIN_Edep_88338887344147
// MI455X (gfx1250) — matrix operand order repaired and checked by reading, not compiled, not run on hardware
//
#include <hip/hip_runtime.h>
#include <hip/hip_bf16.h>

typedef __attribute__((ext_vector_type(16))) _Float16 v16h;
typedef __attribute__((ext_vector_type(8)))  _Float16 v8h;
typedef __attribute__((ext_vector_type(8)))  float    v8f;

#define WAVES_PER_BLOCK 8
#define BLOCK_THREADS   256

// ---------------------------------------------------------------------------
// B fragment loader: W is row-major [KDIM][NDIM] f32 in global memory.
// CDNA5 16-bit B layout (32K x 16N per fragment): lane holds column
// n = nt*16 + lane%16; low lanes K={0..7,16..23}, high lanes K={8..15,24..31}.
// Out-of-range K/N padded with 0 so padded GEMMs are exact.
// ---------------------------------------------------------------------------
template<int KDIM, int NDIM>
__device__ inline v16h load_B_frag(const float* __restrict__ W, int kb, int nt, int lane) {
  const int n  = nt * 16 + (lane & 15);
  const int k0 = kb * 32 + ((lane >> 4) & 1) * 8;
  v16h b;
#pragma unroll
  for (int j = 0; j < 16; ++j) {
    const int k = k0 + (j < 8 ? j : j + 8);
    float v = 0.0f;
    if (k < KDIM && n < NDIM) v = W[k * NDIM + n];
    b[j] = (_Float16)v;
  }
  return b;
}

// ---------------------------------------------------------------------------
// 3-layer MLP (DIN<=32 -> DH -> DH -> DOUT<=16), weights resident in VGPRs.
// forward2() pushes TWO independent 16-row tiles through the MLP. Per layer,
// ALL WMMAs are issued into distinct accumulators first (no WAW reuse, no
// RAW inside the 4-coexec-NOP hazard window), then all epilogues run; the
// K-dim accumulation chains of layer 2/3 are interleaved so same-accumulator
// ops sit >= 2 matrix ops apart (2*NT in layer 2). Activations are transposed through
// per-tile [16][64] f16 LDS tiles (WMMA C-layout -> A-layout).
// ---------------------------------------------------------------------------
template<int DIN, int DH, int DOUT>
struct MlpW {
  static_assert(DIN <= 32 && DOUT <= 16, "tile limits");
  static constexpr int NT = (DH + 15) / 16;   // hidden column tiles (3 for DH=40)
  static constexpr int KB = (DH + 31) / 32;   // hidden K blocks     (2 for DH=40)
  v16h B1[NT];
  v16h B2[NT][KB];
  v16h B3[KB];
  float c1[NT], c2[NT], c3;                   // per-lane bias values

  __device__ void load(const float* W1, const float* b1,
                       const float* W2, const float* b2,
                       const float* W3, const float* b3, int lane) {
    const int n = lane & 15;
#pragma unroll
    for (int nt = 0; nt < NT; ++nt) {
      B1[nt] = load_B_frag<DIN, DH>(W1, 0, nt, lane);
      c1[nt] = (nt * 16 + n < DH) ? b1[nt * 16 + n] : 0.0f;
#pragma unroll
      for (int kb = 0; kb < KB; ++kb)
        B2[nt][kb] = load_B_frag<DH, DH>(W2, kb, nt, lane);
      c2[nt] = (nt * 16 + n < DH) ? b2[nt * 16 + n] : 0.0f;
    }
#pragma unroll
    for (int kb = 0; kb < KB; ++kb)
      B3[kb] = load_B_frag<DH, DOUT>(W3, kb, 0, lane);
    c3 = (n < DOUT) ? b3[n] : 0.0f;
  }

  // bias + ReLU + f16 convert + scatter into row-major LDS (C-layout rows).
  __device__ inline void epilogue(const v8f& h, float bias, _Float16* L,
                                  int nt, int n, int kh) const {
#pragma unroll
    for (int d = 0; d < 8; ++d) {
      float v = h[d] + bias;
      v = v > 0.0f ? v : 0.0f;
      L[(d + kh * 8) * 64 + nt * 16 + n] = (_Float16)v;
    }
  }

  __device__ inline void readA(const _Float16* row, int kh, v16h (&a)[KB]) const {
#pragma unroll
    for (int kb = 0; kb < KB; ++kb) {
      v8h lo = *(const v8h*)(row + kb * 32 + kh * 8);
      v8h hi = *(const v8h*)(row + kb * 32 + 16 + kh * 8);
#pragma unroll
      for (int j = 0; j < 8; ++j) { a[kb][j] = lo[j]; a[kb][8 + j] = hi[j]; }
    }
  }

  __device__ void forward2(v16h a0A, v16h a0B,
                           _Float16* __restrict__ LA, _Float16* __restrict__ LB,
                           int lane, v8f& oA, v8f& oB) const {
    const int n  = lane & 15;
    const int kh = (lane >> 4) & 1;
    _Float16* rowA = LA + n * 64;
    _Float16* rowB = LB + n * 64;
    { v8h z = {}; *(v8h*)(rowA + 48 + kh * 8) = z; *(v8h*)(rowB + 48 + kh * 8) = z; }
    v8f zero = {};
    v8f h[2][NT];   // 6 distinct live accumulators (48 VGPRs) for DH=40

    // ---- layer 1: [16 x DIN] @ [DIN x DH] ----
    // Issue all 2*NT WMMAs back-to-back into distinct accumulators...
#pragma unroll
    for (int nt = 0; nt < NT; ++nt) {
      h[0][nt] = __builtin_amdgcn_wmma_f32_16x16x32_f16(false, a0A, false, B1[nt],
                                                        (short)0, zero, false, false);
      h[1][nt] = __builtin_amdgcn_wmma_f32_16x16x32_f16(false, a0B, false, B1[nt],
                                                        (short)0, zero, false, false);
    }
    // ...then run all epilogues (first producer is 2*NT matrix ops back).
#pragma unroll
    for (int nt = 0; nt < NT; ++nt) {
      epilogue(h[0][nt], c1[nt], LA, nt, n, kh);
      epilogue(h[1][nt], c1[nt], LB, nt, n, kh);
    }
    // gather A fragments for layer 2
    v16h a1A[KB], a1B[KB];
    readA(rowA, kh, a1A);
    readA(rowB, kh, a1B);
    // ---- layer 2: [16 x DH] @ [DH x DH], 2*NT chains interleaved over K ----
#pragma unroll
    for (int nt = 0; nt < NT; ++nt) { h[0][nt] = zero; h[1][nt] = zero; }
#pragma unroll
    for (int kb = 0; kb < KB; ++kb) {
#pragma unroll
      for (int nt = 0; nt < NT; ++nt) {
        h[0][nt] = __builtin_amdgcn_wmma_f32_16x16x32_f16(false, a1A[kb], false, B2[nt][kb],
                                                          (short)0, h[0][nt], false, false);
        h[1][nt] = __builtin_amdgcn_wmma_f32_16x16x32_f16(false, a1B[kb], false, B2[nt][kb],
                                                          (short)0, h[1][nt], false, false);
      }
    }
#pragma unroll
    for (int nt = 0; nt < NT; ++nt) {
      epilogue(h[0][nt], c2[nt], LA, nt, n, kh);
      epilogue(h[1][nt], c2[nt], LB, nt, n, kh);
    }
    // gather A fragments for layer 3
    v16h a2A[KB], a2B[KB];
    readA(rowA, kh, a2A);
    readA(rowB, kh, a2B);
    // ---- layer 3: [16 x DH] @ [DH x DOUT], 2 chains interleaved ----
    v8f tA = zero, tB = zero;
#pragma unroll
    for (int kb = 0; kb < KB; ++kb) {
      tA = __builtin_amdgcn_wmma_f32_16x16x32_f16(false, a2A[kb], false, B3[kb],
                                                  (short)0, tA, false, false);
      tB = __builtin_amdgcn_wmma_f32_16x16x32_f16(false, a2B[kb], false, B3[kb],
                                                  (short)0, tB, false, false);
    }
#pragma unroll
    for (int d = 0; d < 8; ++d) { tA[d] += c3; tB[d] += c3; }
    oA = tA; oB = tB;
  }
};

// ---------------------------------------------------------------------------
// Edge-phase kernel: m1 = concat(x[dst], x[src], edge_attr) -> R1 MLP.
// Processes 32 edges (two 16-row WMMA tiles) per wave iteration.
// Writes e_new (f16 compact) and atomically accumulates segment_sum into
// aggr[dst] (f32, stride 16). FINAL variant runs R2 + sigmoid -> d_out.
// launch_bounds(.,1): relax the VGPR cap so the 6 accumulator groups stay
// in distinct registers (avoids WAW-forced WMMA serialization).
// ---------------------------------------------------------------------------
template<int F, int EF, int DH, int EOUT, typename ET, bool FINAL>
__global__ __launch_bounds__(BLOCK_THREADS, 1)
void edge_mlp_kernel(const float* __restrict__ xnode,
                     const ET*    __restrict__ eattr,
                     const int*   __restrict__ src,
                     const int*   __restrict__ dst,
                     int nEdges,
                     const float* __restrict__ W1, const float* __restrict__ b1,
                     const float* __restrict__ W2, const float* __restrict__ b2,
                     const float* __restrict__ W3, const float* __restrict__ b3,
                     _Float16* __restrict__ eout,
                     float*    __restrict__ aggr,
                     float*    __restrict__ outFinal) {
  constexpr int DIN = 2 * F + EF;
  __shared__ alignas(16) _Float16 lds[WAVES_PER_BLOCK][2][16][64];
  const int lane  = (int)(threadIdx.x & 31);
  const int wslot = (int)(threadIdx.x >> 5);
  MlpW<DIN, DH, EOUT> mlp;
  mlp.load(W1, b1, W2, b2, W3, b3, lane);
  const int n  = lane & 15;
  const int kh = (lane >> 4) & 1;
  const int gwave  = (int)((blockIdx.x * blockDim.x + threadIdx.x) >> 5);
  const int nwave  = (int)((gridDim.x * blockDim.x) >> 5);
  const int npairs = nEdges >> 5;   // E is a multiple of 32

  for (int t = gwave; t < npairs; t += nwave) {
    const int ebase = t << 5;
    // Prefetch next pair's index + edge-attr stream (global_prefetch_b8).
    const int tn = t + nwave;
    if (tn < npairs) {
      const int nb = tn << 5;
      __builtin_prefetch(src + nb + lane, 0, 0);
      __builtin_prefetch(dst + nb + lane, 0, 0);
      __builtin_prefetch(eattr + (size_t)(nb + lane) * EF, 0, 0);
    }

    const int eA = ebase + n;            // tile A gather row
    const int eB = ebase + 16 + n;       // tile B gather row
    const int sA = src[eA], dA = dst[eA];
    const int sB = src[eB], dB = dst[eB];
    const float* xdA = xnode + (long)dA * F;
    const float* xsA = xnode + (long)sA * F;
    const ET*    eaA = eattr + (long)eA * EF;
    const float* xdB = xnode + (long)dB * F;
    const float* xsB = xnode + (long)sB * F;
    const ET*    eaB = eattr + (long)eB * EF;

    auto fetchA = [&](int k) -> _Float16 {   // k is compile-time after unroll
      float v = 0.0f;
      if (k < F)        v = xdA[k];
      else if (k < 2*F) v = xsA[k - F];
      else if (k < DIN) v = (float)eaA[k - 2 * F];
      return (_Float16)v;
    };
    auto fetchB = [&](int k) -> _Float16 {
      float v = 0.0f;
      if (k < F)        v = xdB[k];
      else if (k < 2*F) v = xsB[k - F];
      else if (k < DIN) v = (float)eaB[k - 2 * F];
      return (_Float16)v;
    };
    // CDNA5 16-bit A 16x32 layout: low lanes K={0..7,16..23}, high lanes K={8..15,24..31}
    v16h a0A, a0B;
    if (kh == 0) {
#pragma unroll
      for (int j = 0; j < 8; ++j) {
        a0A[j] = fetchA(j);     a0A[8 + j] = fetchA(16 + j);
        a0B[j] = fetchB(j);     a0B[8 + j] = fetchB(16 + j);
      }
    } else {
#pragma unroll
      for (int j = 0; j < 8; ++j) {
        a0A[j] = fetchA(8 + j); a0A[8 + j] = fetchA(24 + j);
        a0B[j] = fetchB(8 + j); a0B[8 + j] = fetchB(24 + j);
      }
    }

    v8f oA, oB;
    mlp.forward2(a0A, a0B, &lds[wslot][0][0][0], &lds[wslot][1][0][0], lane, oA, oB);

    if constexpr (FINAL) {
      if (n == 0) {                 // EOUT == 1: only column 0 is real
#pragma unroll
        for (int d = 0; d < 8; ++d) {
          const int rA = ebase + d + kh * 8;
          const int rB = rA + 16;
          outFinal[rA] = 1.0f / (1.0f + __expf(-oA[d]));
          outFinal[rB] = 1.0f / (1.0f + __expf(-oB[d]));
        }
      }
    } else {
      if (n < EOUT) {
#pragma unroll
        for (int d = 0; d < 8; ++d) {
          const int rA = ebase + d + kh * 8;
          const int rB = rA + 16;
          eout[(long)rA * EOUT + n] = (_Float16)oA[d];
          eout[(long)rB * EOUT + n] = (_Float16)oB[d];
          atomicAdd(&aggr[(long)dst[rA] * 16 + n], oA[d]);   // segment_sum
          atomicAdd(&aggr[(long)dst[rB] * 16 + n], oB[d]);
        }
      }
    }
  }
}

// ---------------------------------------------------------------------------
// Node-phase kernel: x_new = O_MLP(concat(x, Edep, aggr)); 32 nodes/iter.
// ---------------------------------------------------------------------------
template<int F, int EO, int DH, int FOUT>
__global__ __launch_bounds__(BLOCK_THREADS, 1)
void node_mlp_kernel(const float* __restrict__ xin,
                     const float* __restrict__ Edep,
                     const float* __restrict__ aggr,
                     int nNodes,
                     const float* __restrict__ W1, const float* __restrict__ b1,
                     const float* __restrict__ W2, const float* __restrict__ b2,
                     const float* __restrict__ W3, const float* __restrict__ b3,
                     float* __restrict__ xout) {
  constexpr int DIN = F + 1 + EO;
  __shared__ alignas(16) _Float16 lds[WAVES_PER_BLOCK][2][16][64];
  const int lane  = (int)(threadIdx.x & 31);
  const int wslot = (int)(threadIdx.x >> 5);
  MlpW<DIN, DH, FOUT> mlp;
  mlp.load(W1, b1, W2, b2, W3, b3, lane);
  const int n  = lane & 15;
  const int kh = (lane >> 4) & 1;
  const int gwave  = (int)((blockIdx.x * blockDim.x + threadIdx.x) >> 5);
  const int nwave  = (int)((gridDim.x * blockDim.x) >> 5);
  const int npairs = nNodes >> 5;   // N is a multiple of 32

  for (int t = gwave; t < npairs; t += nwave) {
    const int nbase = t << 5;
    const int nodeA = nbase + n;
    const int nodeB = nbase + 16 + n;
    const float* xrA = xin + (long)nodeA * F;
    const float  edA = Edep[nodeA];
    const float* agA = aggr + (long)nodeA * 16;
    const float* xrB = xin + (long)nodeB * F;
    const float  edB = Edep[nodeB];
    const float* agB = aggr + (long)nodeB * 16;
    auto fetchA = [&](int k) -> _Float16 {
      float v = 0.0f;
      if (k < F)        v = xrA[k];
      else if (k == F)  v = edA;
      else if (k < DIN) v = agA[k - F - 1];
      return (_Float16)v;
    };
    auto fetchB = [&](int k) -> _Float16 {
      float v = 0.0f;
      if (k < F)        v = xrB[k];
      else if (k == F)  v = edB;
      else if (k < DIN) v = agB[k - F - 1];
      return (_Float16)v;
    };
    v16h a0A, a0B;
    if (kh == 0) {
#pragma unroll
      for (int j = 0; j < 8; ++j) {
        a0A[j] = fetchA(j);     a0A[8 + j] = fetchA(16 + j);
        a0B[j] = fetchB(j);     a0B[8 + j] = fetchB(16 + j);
      }
    } else {
#pragma unroll
      for (int j = 0; j < 8; ++j) {
        a0A[j] = fetchA(8 + j); a0A[8 + j] = fetchA(24 + j);
        a0B[j] = fetchB(8 + j); a0B[8 + j] = fetchB(24 + j);
      }
    }

    v8f oA, oB;
    mlp.forward2(a0A, a0B, &lds[wslot][0][0][0], &lds[wslot][1][0][0], lane, oA, oB);

    if (n < FOUT) {
#pragma unroll
      for (int d = 0; d < 8; ++d) {
        const int rA = nbase + d + kh * 8;
        const int rB = rA + 16;
        xout[(long)rA * FOUT + n] = oA[d];
        xout[(long)rB * FOUT + n] = oB[d];
      }
    }
  }
}

__global__ void zero_f32_kernel(float* __restrict__ p, int count) {
  int i = (int)(blockIdx.x * blockDim.x + threadIdx.x);
  if (i < count) p[i] = 0.0f;
}

// ---------------------------------------------------------------------------
// Host-side orchestration.
// d_in order: x, Edep, edge_index, edge_attr, then params flattened in dict
// order: IN1_R1{W1,b1,W2,b2,W3,b3}, IN1_O, IN2_R1, IN2_O, IN3_R1, IN3_O, R2.
// ---------------------------------------------------------------------------
extern "C" void kernel_launch(void* const* d_in, const int* in_sizes, int n_in,
                              void* d_out, int out_size, void* d_ws, size_t ws_size,
                              hipStream_t stream) {
  const float* x      = (const float*)d_in[0];
  const float* Edep   = (const float*)d_in[1];
  const int*   eidx   = (const int*)  d_in[2];
  const float* eattr0 = (const float*)d_in[3];
  const int N = in_sizes[0] / 3;      // 100000
  const int E = in_sizes[2] / 2;      // 3200000
  const int* src = eidx;              // edge_index[0]
  const int* dst = eidx + E;          // edge_index[1]
  auto P = [&](int i) { return (const float*)d_in[i]; };

  // Workspace carve-out (~163 MB)
  char* w = (char*)d_ws;
  auto carve = [&](size_t bytes) { void* p = (void*)w; w += (bytes + 255) & ~(size_t)255; return p; };
  float*     x1   = (float*)    carve((size_t)N * 6 * sizeof(float));
  float*     x2   = (float*)    carve((size_t)N * 8 * sizeof(float));
  float*     x3   = (float*)    carve((size_t)N * 9 * sizeof(float));
  float*     aggr = (float*)    carve((size_t)N * 16 * sizeof(float));
  _Float16*  e1   = (_Float16*) carve((size_t)E * 6 * sizeof(_Float16));
  _Float16*  e2   = (_Float16*) carve((size_t)E * 8 * sizeof(_Float16));
  _Float16*  e3   = (_Float16*) carve((size_t)E * 9 * sizeof(_Float16));
  float* out = (float*)d_out;

  const dim3 blk(BLOCK_THREADS);
  const int edgeGrid = 1024;                       // 8192 persistent waves, ~12 pairs each
  const int nodeGrid = 128;                        // 1024 persistent waves
  const int zGrid    = (N * 16 + BLOCK_THREADS - 1) / BLOCK_THREADS;

  // ---- IN layer 1: node 3->6, edge 3->6 ----
  zero_f32_kernel<<<zGrid, blk, 0, stream>>>(aggr, N * 16);
  edge_mlp_kernel<3, 3, 40, 6, float, false><<<edgeGrid, blk, 0, stream>>>(
      x, eattr0, src, dst, E, P(4), P(5), P(6), P(7), P(8), P(9), e1, aggr, nullptr);
  node_mlp_kernel<3, 6, 40, 6><<<nodeGrid, blk, 0, stream>>>(
      x, Edep, aggr, N, P(10), P(11), P(12), P(13), P(14), P(15), x1);

  // ---- IN layer 2: node 6->8, edge 6->8 ----
  zero_f32_kernel<<<zGrid, blk, 0, stream>>>(aggr, N * 16);
  edge_mlp_kernel<6, 6, 40, 8, _Float16, false><<<edgeGrid, blk, 0, stream>>>(
      x1, e1, src, dst, E, P(16), P(17), P(18), P(19), P(20), P(21), e2, aggr, nullptr);
  node_mlp_kernel<6, 8, 40, 8><<<nodeGrid, blk, 0, stream>>>(
      x1, Edep, aggr, N, P(22), P(23), P(24), P(25), P(26), P(27), x2);

  // ---- IN layer 3: node 8->9, edge 8->9 ----
  zero_f32_kernel<<<zGrid, blk, 0, stream>>>(aggr, N * 16);
  edge_mlp_kernel<8, 8, 40, 9, _Float16, false><<<edgeGrid, blk, 0, stream>>>(
      x2, e2, src, dst, E, P(28), P(29), P(30), P(31), P(32), P(33), e3, aggr, nullptr);
  node_mlp_kernel<8, 9, 40, 9><<<nodeGrid, blk, 0, stream>>>(
      x2, Edep, aggr, N, P(34), P(35), P(36), P(37), P(38), P(39), x3);

  // ---- Final R2: edge 27 -> 40 -> 40 -> 1, sigmoid ----
  edge_mlp_kernel<9, 9, 40, 1, _Float16, true><<<edgeGrid, blk, 0, stream>>>(
      x3, e3, src, dst, E, P(40), P(41), P(42), P(43), P(44), P(45), nullptr, nullptr, out);
}
